// AudioEncoder_24352464570130
// MI455X (gfx1250) — matrix operand order repaired and checked by reading, not compiled, not run on hardware
//
#include <hip/hip_runtime.h>

// CDNA5 (gfx1250) wave32 WMMA LSTM, warp-specialized (waves 0-3: layer 0, waves 4-7: layer 1).
typedef __attribute__((ext_vector_type(16))) _Float16 v16h;
typedef __attribute__((ext_vector_type(8)))  float    v8f;

#define LSTM_H 64   // hidden size (hard-coded in reference)
#define BT     16   // batch-tile rows per workgroup
#define XC     64   // x staging chunk (timesteps)

__device__ __forceinline__ float fast_sig(float x) {
    return __builtin_amdgcn_rcpf(1.0f + __expf(-x));
}
__device__ __forceinline__ float fast_tanh(float x) {
    return 2.0f * __builtin_amdgcn_rcpf(1.0f + __expf(-2.0f * x)) - 1.0f;
}

// A fragment (16x32 f16) from LDS, h stored row-major [16][LSTM_H].
// ISA layout (16-bit A 16x32): lane m=lane&15, g=lane>>4;
// VGPR r<4 : K = k0 + 8g + 2r + j ; VGPR r>=4 : K = k0 + 8g + 16 + 2(r-4) + j
__device__ __forceinline__ v16h load_a_frag(const _Float16* hbuf, int lane, int k0) {
    const int m = lane & 15, g = lane >> 4;
    const _Float16* p = hbuf + m * LSTM_H + k0 + 8 * g;
    v16h a;
#pragma unroll
    for (int r = 0; r < 8; ++r) {
        const int kk = ((r >= 4) ? 16 : 0) + 2 * (r & 3);
        a[2 * r]     = p[kk];
        a[2 * r + 1] = p[kk + 1];
    }
    return a;
}

// B fragment (32x16 f16) = W^T tile. W is [N][K] f32 row-major, K stride = LSTM_H.
// Layout: N = n0 + (lane&15); K = k0 + 8*(lane>>4) + idx (idx<8), k0 + 16 + 8*(lane>>4) + idx-8 (idx>=8).
__device__ __forceinline__ v16h load_b_frag(const float* __restrict__ W, int lane, int n0, int k0) {
    const int n = n0 + (lane & 15), g = lane >> 4;
    const float* p = W + n * LSTM_H + k0 + 8 * g;
    v16h b;
#pragma unroll
    for (int i = 0; i < 16; ++i) b[i] = (_Float16)p[(i < 8) ? i : i + 8];
    return b;
}

#define WMMA_F16(A, B, C) \
    __builtin_amdgcn_wmma_f32_16x16x32_f16(false, (A), false, (B), (short)0, (C), false, false)

__global__ void __launch_bounds__(256, 1)
lstm2_fc_wmma_kernel(const float* __restrict__ x,
                     const float* __restrict__ W_ih0, const float* __restrict__ W_hh0,
                     const float* __restrict__ b_ih0, const float* __restrict__ b_hh0,
                     const float* __restrict__ W_ih1, const float* __restrict__ W_hh1,
                     const float* __restrict__ b_ih1, const float* __restrict__ b_hh1,
                     const float* __restrict__ W_fc,  const float* __restrict__ b_fc,
                     float* __restrict__ out, int T, int E)
{
    __shared__ _Float16 h0buf[2][BT * LSTM_H];   // layer-0 hidden, double buffered (f16)
    __shared__ _Float16 h1buf[2][BT * LSTM_H];   // layer-1 hidden, double buffered (f16)
    __shared__ float    xbuf[XC * BT];           // staged x chunk, [t_local][row]

    const int tid   = threadIdx.x;
    const int lane  = tid & 31;
    const int wave  = tid >> 5;          // 0..7 : waves 0-3 = layer0, 4-7 = layer1
    const int g     = lane >> 4;         // lane half
    const int nn    = lane & 15;         // column within 16-wide tile
    const int bbase = blockIdx.x * BT;
    const int u0    = (wave & 3) * 16;   // this wave's 16-wide u-slice of H

    // ---- zero-init hidden-state LDS buffers ----
    {
        _Float16* p0 = &h0buf[0][0];
        _Float16* p1 = &h1buf[0][0];
        for (int i = tid; i < 2 * BT * LSTM_H; i += 256) { p0[i] = (_Float16)0.0f; p1[i] = (_Float16)0.0f; }
    }
    __syncthreads();

    if (wave < 4) {
        // ================= layer-0 producer waves =================
        v16h Bhh0[4][2];
        float bias0[4], xw0[4];
#pragma unroll
        for (int gi = 0; gi < 4; ++gi) {
            const int col = gi * LSTM_H + u0;
#pragma unroll
            for (int kc = 0; kc < 2; ++kc)
                Bhh0[gi][kc] = load_b_frag(W_hh0, lane, col, kc * 32);
            bias0[gi] = b_ih0[col + nn] + b_hh0[col + nn];
            xw0[gi]   = W_ih0[col + nn];           // W_ih0 is [4H][1]
        }
        v8f cst;
#pragma unroll
        for (int r = 0; r < 8; ++r) cst[r] = 0.0f;

        for (int it = 0; it <= T; ++it) {
            const int prev = (it + 1) & 1;
            const int cur  = it & 1;
            const bool stage = ((it & (XC - 1)) == 0) && (it < T);
            if (stage) {
                // 128 threads: stage x[t0 .. t0+XC) transposed to [t][row]
                const int t0  = it;
                const int row = tid & 15;
                const int c8  = (tid >> 4) * 8;    // tid>>4 in 0..7
                const float* xp = x + (size_t)(bbase + row) * T + t0 + c8;
                if (t0 + c8 + 7 < T) {
                    float4 va = *(const float4*)xp;
                    float4 vb = *(const float4*)(xp + 4);
                    xbuf[(c8 + 0) * BT + row] = va.x;
                    xbuf[(c8 + 1) * BT + row] = va.y;
                    xbuf[(c8 + 2) * BT + row] = va.z;
                    xbuf[(c8 + 3) * BT + row] = va.w;
                    xbuf[(c8 + 4) * BT + row] = vb.x;
                    xbuf[(c8 + 5) * BT + row] = vb.y;
                    xbuf[(c8 + 6) * BT + row] = vb.z;
                    xbuf[(c8 + 7) * BT + row] = vb.w;
                } else {
#pragma unroll
                    for (int j = 0; j < 8; ++j)
                        xbuf[(c8 + j) * BT + row] = (t0 + c8 + j < T) ? xp[j] : 0.0f;
                }
                if (t0 + XC < T) __builtin_prefetch(xp + XC, 0, 1);   // global_prefetch next chunk
            }
            if (stage) __syncthreads();

            if (it < T) {
                const int tl = it & (XC - 1);
                const float* xr = xbuf + tl * BT + 8 * g;
                float xm[8];
                {
                    float4 va = *(const float4*)(xr);
                    float4 vb = *(const float4*)(xr + 4);
                    xm[0]=va.x; xm[1]=va.y; xm[2]=va.z; xm[3]=va.w;
                    xm[4]=vb.x; xm[5]=vb.y; xm[6]=vb.z; xm[7]=vb.w;
                }
                v16h a0 = load_a_frag(h0buf[prev], lane, 0);
                v16h a1 = load_a_frag(h0buf[prev], lane, 32);
                v8f gacc[4];
#pragma unroll
                for (int gi = 0; gi < 4; ++gi) {
                    v8f acc;
#pragma unroll
                    for (int r = 0; r < 8; ++r) acc[r] = fmaf(xm[r], xw0[gi], bias0[gi]);
                    acc = WMMA_F16(a0, Bhh0[gi][0], acc);
                    acc = WMMA_F16(a1, Bhh0[gi][1], acc);
                    gacc[gi] = acc;
                }
#pragma unroll
                for (int r = 0; r < 8; ++r) {
                    const float ig = fast_sig(gacc[0][r]);
                    const float fg = fast_sig(gacc[1][r]);
                    const float gg = fast_tanh(gacc[2][r]);
                    const float og = fast_sig(gacc[3][r]);
                    const float cn = fmaf(fg, cst[r], ig * gg);
                    cst[r] = cn;
                    h0buf[cur][(r + 8 * g) * LSTM_H + u0 + nn] = (_Float16)(og * fast_tanh(cn));
                }
            }
            __syncthreads();
        }
    } else {
        // ================= layer-1 consumer waves (one step behind) =================
        v16h Bih1[4][2], Bhh1[4][2];
        float bias1[4];
#pragma unroll
        for (int gi = 0; gi < 4; ++gi) {
            const int col = gi * LSTM_H + u0;
#pragma unroll
            for (int kc = 0; kc < 2; ++kc) {
                Bih1[gi][kc] = load_b_frag(W_ih1, lane, col, kc * 32);
                Bhh1[gi][kc] = load_b_frag(W_hh1, lane, col, kc * 32);
            }
            bias1[gi] = b_ih1[col + nn] + b_hh1[col + nn];
        }
        v8f cst;
#pragma unroll
        for (int r = 0; r < 8; ++r) cst[r] = 0.0f;

        for (int it = 0; it <= T; ++it) {
            const int prev = (it + 1) & 1;
            const int cur  = it & 1;
            const bool stage = ((it & (XC - 1)) == 0) && (it < T);
            if (stage) __syncthreads();        // match producer barrier pattern

            if (it >= 1) {
                v16h a00 = load_a_frag(h0buf[prev], lane, 0);   // layer-0 output (step it-1)
                v16h a01 = load_a_frag(h0buf[prev], lane, 32);
                v16h a10 = load_a_frag(h1buf[prev], lane, 0);   // own previous hidden
                v16h a11 = load_a_frag(h1buf[prev], lane, 32);
                v8f gacc[4];
#pragma unroll
                for (int gi = 0; gi < 4; ++gi) {
                    v8f acc;
#pragma unroll
                    for (int r = 0; r < 8; ++r) acc[r] = bias1[gi];
                    acc = WMMA_F16(a00, Bih1[gi][0], acc);
                    acc = WMMA_F16(a01, Bih1[gi][1], acc);
                    acc = WMMA_F16(a10, Bhh1[gi][0], acc);
                    acc = WMMA_F16(a11, Bhh1[gi][1], acc);
                    gacc[gi] = acc;
                }
#pragma unroll
                for (int r = 0; r < 8; ++r) {
                    const float ig = fast_sig(gacc[0][r]);
                    const float fg = fast_sig(gacc[1][r]);
                    const float gg = fast_tanh(gacc[2][r]);
                    const float og = fast_sig(gacc[3][r]);
                    const float cn = fmaf(fg, cst[r], ig * gg);
                    cst[r] = cn;
                    h1buf[cur][(r + 8 * g) * LSTM_H + u0 + nn] = (_Float16)(og * fast_tanh(cn));
                }
            }
            __syncthreads();
        }
    }

    // ---- final FC: out[b, e] = h1_last @ W_fc^T + b_fc ; [16 x 64] x [64 x E] ----
    const int fbuf = T & 1;               // buffer holding h1 of step T-1
    v16h af0 = load_a_frag(h1buf[fbuf], lane, 0);
    v16h af1 = load_a_frag(h1buf[fbuf], lane, 32);
    for (int n0 = wave * 16; n0 < E; n0 += 16 * 8) {
        v16h bf0 = load_b_frag(W_fc, lane, n0, 0);
        v16h bf1 = load_b_frag(W_fc, lane, n0, 32);
        const float bb = b_fc[n0 + nn];
        v8f acc;
#pragma unroll
        for (int r = 0; r < 8; ++r) acc[r] = bb;
        acc = WMMA_F16(af0, bf0, acc);
        acc = WMMA_F16(af1, bf1, acc);
#pragma unroll
        for (int r = 0; r < 8; ++r)
            out[(size_t)(bbase + r + 8 * g) * E + n0 + nn] = acc[r];
    }
}

extern "C" void kernel_launch(void* const* d_in, const int* in_sizes, int n_in,
                              void* d_out, int out_size, void* d_ws, size_t ws_size,
                              hipStream_t stream) {
    (void)n_in; (void)d_ws; (void)ws_size;
    const float* x     = (const float*)d_in[0];
    const float* W_ih0 = (const float*)d_in[1];
    const float* W_hh0 = (const float*)d_in[2];
    const float* b_ih0 = (const float*)d_in[3];
    const float* b_hh0 = (const float*)d_in[4];
    const float* W_ih1 = (const float*)d_in[5];
    const float* W_hh1 = (const float*)d_in[6];
    const float* b_ih1 = (const float*)d_in[7];
    const float* b_hh1 = (const float*)d_in[8];
    const float* W_fc  = (const float*)d_in[9];
    const float* b_fc  = (const float*)d_in[10];
    float* out = (float*)d_out;

    const int E = in_sizes[9] / LSTM_H;      // W_fc is [E, H]
    const int B = out_size / E;              // out is [B, E]
    const int T = in_sizes[0] / B;           // x is [B, T, 1]

    dim3 grid(B / BT), block(256);
    hipLaunchKernelGGL(lstm2_fc_wmma_kernel, grid, block, 0, stream,
                       x, W_ih0, W_hh0, b_ih0, b_hh0,
                       W_ih1, W_hh1, b_ih1, b_hh1, W_fc, b_fc,
                       out, T, E);
}
